// Attention_21371757265022
// MI455X (gfx1250) — matrix operand order repaired and checked by reading, not compiled, not run on hardware
//
#include <hip/hip_runtime.h>
#include <hip/hip_bf16.h>

typedef __bf16 bf16_t;
typedef __attribute__((ext_vector_type(8)))  __bf16 bf16x8;
typedef __attribute__((ext_vector_type(16))) __bf16 bf16x16;
typedef __attribute__((ext_vector_type(8)))  float  f32x8;

#define BM    128
#define BN    128
#define KTILE 32
#define PITCH 40   // 32 data halves + 8 pad halves = 80B pitch: 16B aligned, conflict-free

// CDNA5 async global->LDS copy (ASYNCcnt pipeline). LDS dest = low 32 bits of
// the flat address (LDS aperture keeps byte offset in addr[31:0]).
__device__ __forceinline__ void async_ld_b128(void* lds, const void* g)
{
    const unsigned int       l  = (unsigned int)(unsigned long long)(uintptr_t)lds;
    const unsigned long long ga = (unsigned long long)(uintptr_t)g;
    asm volatile("global_load_async_to_lds_b128 %0, %1, off"
                 :: "v"(l), "v"(ga) : "memory");
}
__device__ __forceinline__ void wait_async_4() { asm volatile("s_wait_asynccnt 0x4" ::: "memory"); }
__device__ __forceinline__ void wait_async_0() { asm volatile("s_wait_asynccnt 0x0" ::: "memory"); }

// ---------------------------------------------------------------------------
// Generic "NT" GEMM: C[m][n] = sum_k A[m*lda + k] * B[n*ldb + k]
// bf16 inputs, f32 WMMA accumulation, 128x128 block tile, 8 wave32 waves
// (each wave: 64x32 = 4x2 WMMA tiles), double-buffered LDS fed by
// global_load_async_to_lds_b128; last K-stage peeled so the steady-state loop
// is branch-free. Epilogue selected by EPI:
//   0: outF[m*N + n] = c                                   (final output)
//   1: outB[((m>>seqShift)*N + n) << seqShift | m&mask]    (q/k: transposed (B,D,A) bf16)
//   2: outB[m*N + n] = bf16(c)                             (v: (B,A,D) bf16)
//   3: outB[bz*M*N + n*M + m] = bf16(c)                    (ctx: transpose back to (B,A,D))
//   4: outF[bz*M*N + m*N + n] = c*invA[m]*invB[n]*temp[bz] (attn logits, norms folded in)
// ---------------------------------------------------------------------------
template<int EPI>
__global__ __launch_bounds__(256) void gemm_bf16_nt(
    const bf16_t* __restrict__ Aop, const bf16_t* __restrict__ Bop,
    int M, int N, int K, int lda, int ldb,
    long long bsA, long long bsB,
    float* __restrict__ outF, bf16_t* __restrict__ outB,
    int seqShift,
    const float* __restrict__ invA, const float* __restrict__ invB,
    const float* __restrict__ temp)
{
    __shared__ alignas(16) bf16_t As[2][BM][PITCH];
    __shared__ alignas(16) bf16_t Bs[2][BN][PITCH];

    const int bz = blockIdx.z;
    const bf16_t* Ab = Aop + (long long)bz * bsA;
    const bf16_t* Bb = Bop + (long long)bz * bsB;

    const int tid  = threadIdx.x;      // 0..255  (8 wave32 waves)
    const int lane = tid & 31;
    const int wave = tid >> 5;
    const int lm   = lane & 15;        // row/col within 16x16 tile
    const int hi   = lane >> 4;        // lane-group (K half split per ISA layout)
    const int wrow = (wave >> 2) * 64; // wave tile origin inside 128x128 block tile
    const int wcol = (wave & 3) * 32;
    const int m0 = blockIdx.y * BM;
    const int n0 = blockIdx.x * BN;

    // global->LDS copy map: 128 rows x 32 halves per tile; thread t copies 16
    // halves (two b128 async loads) of A and of B per stage.
    const int crow = tid >> 1;
    const int ccol = (tid & 1) * 16;

    f32x8 acc[4][2] = {};

    auto issue = [&](int s, int kk) {
        const bf16_t* sa = Ab + (size_t)(m0 + crow) * lda + (kk + ccol);
        const bf16_t* sb = Bb + (size_t)(n0 + crow) * ldb + (kk + ccol);
        async_ld_b128(&As[s][crow][ccol],     sa);
        async_ld_b128(&As[s][crow][ccol + 8], sa + 8);
        async_ld_b128(&Bs[s][crow][ccol],     sb);
        async_ld_b128(&Bs[s][crow][ccol + 8], sb + 8);
    };

    auto compute = [&](int s) {
        // A fragment (16x32): lane lm=row; low lanes hold K 0-7,16-23; high 8-15,24-31
        // B fragment (32x16): lane lm=col; low lanes hold K 0-7,16-23; high 8-15,24-31
        bf16x16 af[4], bfr[2];
        #pragma unroll
        for (int i = 0; i < 4; ++i) {
            const bf16_t* ap = &As[s][wrow + i * 16 + lm][hi * 8];
            const bf16x8 lo  = *(const bf16x8*)ap;
            const bf16x8 hh  = *(const bf16x8*)(ap + 16);
            af[i] = __builtin_shufflevector(lo, hh, 0,1,2,3,4,5,6,7,8,9,10,11,12,13,14,15);
        }
        #pragma unroll
        for (int j = 0; j < 2; ++j) {
            const bf16_t* bp = &Bs[s][wcol + j * 16 + lm][hi * 8];
            const bf16x8 bl  = *(const bf16x8*)bp;
            const bf16x8 bh  = *(const bf16x8*)(bp + 16);
            bfr[j] = __builtin_shufflevector(bl, bh, 0,1,2,3,4,5,6,7,8,9,10,11,12,13,14,15);
        }
        #pragma unroll
        for (int i = 0; i < 4; ++i)
            #pragma unroll
            for (int j = 0; j < 2; ++j)
                acc[i][j] = __builtin_amdgcn_wmma_f32_16x16x32_bf16(
                    false, af[i], false, bfr[j], (short)0, acc[i][j], false, false);
    };

    issue(0, 0);
    int cur = 0;
    // steady state: branch-free — prefetch next stage, wait oldest 4, compute
    for (int kk = 0; kk + KTILE < K; kk += KTILE) {
        issue(cur ^ 1, kk + KTILE);   // idle buffer: all readers passed last barrier
        wait_async_4();               // own 4 loads of current buffer (oldest) done
        __syncthreads();              // every wave's current-buffer data landed
        compute(cur);
        __syncthreads();              // readers done: buffer may be re-filled
        cur ^= 1;
    }
    // peeled last stage
    wait_async_0();
    __syncthreads();
    compute(cur);

    // Epilogue: C 16x16 f32 layout — reg r, lanes0-15: (M=r, N=lane); lanes16-31: (M=8+r)
    #pragma unroll
    for (int i = 0; i < 4; ++i)
        #pragma unroll
        for (int j = 0; j < 2; ++j) {
            const int mbase = m0 + wrow + i * 16 + hi * 8;
            const int n     = n0 + wcol + j * 16 + lm;
            #pragma unroll
            for (int r = 0; r < 8; ++r) {
                const int m   = mbase + r;
                const float c = acc[i][j][r];
                if (EPI == 0) {
                    outF[(size_t)m * N + n] = c;
                } else if (EPI == 1) {
                    const int b = m >> seqShift;
                    const int a = m & ((1 << seqShift) - 1);
                    outB[(((size_t)b * N + n) << seqShift) + a] = (bf16_t)c;
                } else if (EPI == 2) {
                    outB[(size_t)m * N + n] = (bf16_t)c;
                } else if (EPI == 3) {
                    outB[(size_t)bz * M * N + (size_t)n * M + m] = (bf16_t)c;
                } else { // EPI == 4
                    const float s = invA[bz * M + m] * invB[bz * N + n] * temp[bz];
                    outF[(size_t)bz * M * N + (size_t)m * N + n] = c * s;
                }
            }
        }
}

// ---------------------------------------------------------------------------
// inv_norm[row] = 1 / max(||row||_2, eps) over len bf16 elements
// ---------------------------------------------------------------------------
__global__ __launch_bounds__(256) void rownorm_inv(
    const bf16_t* __restrict__ q, float* __restrict__ inv, const int len)
{
    __shared__ float red[8];
    const size_t row = blockIdx.x;
    const bf16_t* p = q + row * (size_t)len;
    float s = 0.f;
    for (int i = threadIdx.x; i < len; i += 256) {
        const float v = (float)p[i];
        s += v * v;
    }
    #pragma unroll
    for (int off = 16; off; off >>= 1) s += __shfl_xor(s, off, 32);
    if ((threadIdx.x & 31) == 0) red[threadIdx.x >> 5] = s;
    __syncthreads();
    if (threadIdx.x == 0) {
        float t = 0.f;
        #pragma unroll
        for (int w = 0; w < 8; ++w) t += red[w];
        inv[row] = 1.0f / fmaxf(sqrtf(t), 1e-12f);
    }
}

// ---------------------------------------------------------------------------
// row softmax: f32 in -> bf16 out
// ---------------------------------------------------------------------------
__global__ __launch_bounds__(256) void softmax_row(
    const float* __restrict__ in, bf16_t* __restrict__ out, const int len)
{
    __shared__ float red[8];
    const size_t row = blockIdx.x;
    const float* p = in + row * (size_t)len;
    bf16_t* o = out + row * (size_t)len;

    float mx = -3.402823466e38f;
    for (int i = threadIdx.x; i < len; i += 256) mx = fmaxf(mx, p[i]);
    #pragma unroll
    for (int off = 16; off; off >>= 1) mx = fmaxf(mx, __shfl_xor(mx, off, 32));
    if ((threadIdx.x & 31) == 0) red[threadIdx.x >> 5] = mx;
    __syncthreads();
    mx = red[0];
    #pragma unroll
    for (int w = 1; w < 8; ++w) mx = fmaxf(mx, red[w]);
    __syncthreads();

    float sum = 0.f;
    for (int i = threadIdx.x; i < len; i += 256) sum += __expf(p[i] - mx);
    #pragma unroll
    for (int off = 16; off; off >>= 1) sum += __shfl_xor(sum, off, 32);
    if ((threadIdx.x & 31) == 0) red[threadIdx.x >> 5] = sum;
    __syncthreads();
    sum = 0.f;
    #pragma unroll
    for (int w = 0; w < 8; ++w) sum += red[w];
    const float invs = 1.0f / sum;
    for (int i = threadIdx.x; i < len; i += 256)
        o[i] = (bf16_t)(__expf(p[i] - mx) * invs);
}

// ---------------------------------------------------------------------------
// fp32 -> bf16 convert, 4 elements/thread (all sizes are multiples of 4)
// ---------------------------------------------------------------------------
__global__ __launch_bounds__(256) void cvt_f32_bf16(
    const float* __restrict__ in, bf16_t* __restrict__ out, const size_t n)
{
    const size_t i = ((size_t)blockIdx.x * 256 + threadIdx.x) * 4;
    if (i + 3 < n) {
        const float4 v = *(const float4*)(in + i);
        out[i + 0] = (bf16_t)v.x;
        out[i + 1] = (bf16_t)v.y;
        out[i + 2] = (bf16_t)v.z;
        out[i + 3] = (bf16_t)v.w;
    }
}

// ---------------------------------------------------------------------------
extern "C" void kernel_launch(void* const* d_in, const int* in_sizes, int n_in,
                              void* d_out, int out_size, void* d_ws, size_t ws_size,
                              hipStream_t stream)
{
    (void)in_sizes; (void)n_in; (void)out_size; (void)ws_size;
    const float* x    = (const float*)d_in[0];
    const float* Wq   = (const float*)d_in[1];
    const float* Wk   = (const float*)d_in[2];
    const float* Wv   = (const float*)d_in[3];
    const float* Wo   = (const float*)d_in[4];
    const float* temp = (const float*)d_in[5];

    const int B = 8, A = 2048, D = 1024;
    const int SEQSHIFT = 11;                 // A == 2048
    const size_t MA = (size_t)B * A;         // 16384 rows for projection GEMMs

    char* p = (char*)d_ws;
    auto take = [&](size_t bytes) -> char* {
        char* r = p;
        p += (bytes + 255) & ~(size_t)255;
        return r;
    };
    bf16_t* x_bf    = (bf16_t*)take(MA * D * 2);          // reused for ctx later
    bf16_t* Wq_bf   = (bf16_t*)take((size_t)D * D * 2);
    bf16_t* Wk_bf   = (bf16_t*)take((size_t)D * D * 2);
    bf16_t* Wv_bf   = (bf16_t*)take((size_t)D * D * 2);
    bf16_t* Wo_bf   = (bf16_t*)take((size_t)D * D * 2);
    bf16_t* q_bf    = (bf16_t*)take((size_t)B * D * A * 2);   // (B,D,A)
    bf16_t* k_bf    = (bf16_t*)take((size_t)B * D * A * 2);   // (B,D,A)
    bf16_t* v_bf    = (bf16_t*)take((size_t)B * A * D * 2);   // (B,A,D)
    float*  attn_f  = (float*) take((size_t)B * D * D * 4);   // (B,D,D)
    bf16_t* attn_bf = (bf16_t*)take((size_t)B * D * D * 2);
    float*  invq    = (float*) take((size_t)B * D * 4);
    float*  invk    = (float*) take((size_t)B * D * 4);
    bf16_t* ctx_bf  = x_bf;   // x dead after q/k/v GEMMs; reuse its 32MB for ctx

    // 1) convert operands to bf16
    cvt_f32_bf16<<<dim3((unsigned)(MA * D / 1024)), 256, 0, stream>>>(x, x_bf, MA * D);
    cvt_f32_bf16<<<dim3((unsigned)((size_t)D * D / 1024)), 256, 0, stream>>>(Wq, Wq_bf, (size_t)D * D);
    cvt_f32_bf16<<<dim3((unsigned)((size_t)D * D / 1024)), 256, 0, stream>>>(Wk, Wk_bf, (size_t)D * D);
    cvt_f32_bf16<<<dim3((unsigned)((size_t)D * D / 1024)), 256, 0, stream>>>(Wv, Wv_bf, (size_t)D * D);
    cvt_f32_bf16<<<dim3((unsigned)((size_t)D * D / 1024)), 256, 0, stream>>>(Wo, Wo_bf, (size_t)D * D);

    // 2) q/k/v projections: C = x @ W.T (rows of W are already K-contiguous B operands)
    const dim3 gProj(D / BN, (unsigned)(MA / BM), 1);         // (8, 128, 1)
    gemm_bf16_nt<1><<<gProj, 256, 0, stream>>>(x_bf, Wq_bf, (int)MA, D, D, D, D, 0, 0,
                                               nullptr, q_bf, SEQSHIFT, nullptr, nullptr, nullptr);
    gemm_bf16_nt<1><<<gProj, 256, 0, stream>>>(x_bf, Wk_bf, (int)MA, D, D, D, D, 0, 0,
                                               nullptr, k_bf, SEQSHIFT, nullptr, nullptr, nullptr);
    gemm_bf16_nt<2><<<gProj, 256, 0, stream>>>(x_bf, Wv_bf, (int)MA, D, D, D, D, 0, 0,
                                               nullptr, v_bf, 0, nullptr, nullptr, nullptr);

    // 3) L2 norms over the token axis (folded into attn GEMM as scales)
    rownorm_inv<<<dim3(B * D), 256, 0, stream>>>(q_bf, invq, A);
    rownorm_inv<<<dim3(B * D), 256, 0, stream>>>(k_bf, invk, A);

    // 4) attn logits: per-batch (D x D), K = A tokens; scale by invq*invk*temperature
    const dim3 gAttn(D / BN, D / BM, B);                      // (8, 8, 8)
    gemm_bf16_nt<4><<<gAttn, 256, 0, stream>>>(q_bf, k_bf, D, D, A, A, A,
                                               (long long)D * A, (long long)D * A,
                                               attn_f, nullptr, 0, invq, invk, temp);

    // 5) softmax over last axis
    softmax_row<<<dim3(B * D), 256, 0, stream>>>(attn_f, attn_bf, D);

    // 6) ctx = attn @ v : per-batch M=D rows(d), N=A cols(a), K=D(e); store back as (B,A,D)
    const dim3 gCtx(A / BN, D / BM, B);                       // (16, 8, 8)
    gemm_bf16_nt<3><<<gCtx, 256, 0, stream>>>(attn_bf, v_bf, D, A, D, D, D,
                                              (long long)D * D, (long long)A * D,
                                              nullptr, ctx_bf, 0, nullptr, nullptr, nullptr);

    // 7) output projection: d_out = ctx @ Wo.T (fp32 out)
    gemm_bf16_nt<0><<<gProj, 256, 0, stream>>>(ctx_bf, Wo_bf, (int)MA, D, D, D, D, 0, 0,
                                               (float*)d_out, nullptr, 0, nullptr, nullptr, nullptr);
}
